// tscnet_9560597200925
// MI455X (gfx1250) — matrix operand order repaired and checked by reading, not compiled, not run on hardware
//
#include <hip/hip_runtime.h>
#include <math.h>

typedef __attribute__((ext_vector_type(16))) _Float16 v16h;
typedef __attribute__((ext_vector_type(8)))  float    v8f;

#define DF __device__ __forceinline__
static inline int cdiv(int a, int b) { return (a + b - 1) / b; }

DF float frcp(float x) { return __builtin_amdgcn_rcpf(x); }          // v_rcp_f32
DF float sigmf(float x) { return frcp(1.f + __expf(-x)); }           // fast sigmoid

// ---------------- model dims ----------------
static constexpr int BN = 2, TT = 321, FQ = 201, FE = 101, F2 = 202;
static constexpr int N1 = BN * TT * FQ;   // 129042 tokens (full freq)
static constexpr int N2 = BN * TT * FE;   // 64842 tokens (encoded)
static constexpr int N3 = BN * TT * F2;   // 129684 tokens (after subpixel)

// =====================================================================
//                              KERNELS
// =====================================================================

__global__ void k_prep(const float* __restrict__ xr, const float* __restrict__ xi,
                       float* __restrict__ x3, float* __restrict__ mag, int n) {
  int i = blockIdx.x * blockDim.x + threadIdx.x; if (i >= n) return;
  float r = xr[i], im = xi[i];
  float m = sqrtf(r * r + im * im);
  mag[i] = m;
  x3[3 * i + 0] = m; x3[3 * i + 1] = r; x3[3 * i + 2] = im;
}

__global__ void k_copy(float* __restrict__ d, const float* __restrict__ s, int n) {
  int i = blockIdx.x * blockDim.x + threadIdx.x; if (i >= n) return; d[i] = s[i];
}
__global__ void k_axpy(float* __restrict__ d, const float* __restrict__ s, float a, int n) {
  int i = blockIdx.x * blockDim.x + threadIdx.x; if (i >= n) return; d[i] += a * s[i];
}
__global__ void k_add2(float* __restrict__ d, const float* __restrict__ a,
                       const float* __restrict__ b, int n) {
  int i = blockIdx.x * blockDim.x + threadIdx.x; if (i >= n) return; d[i] = a[i] + b[i];
}
// copy [tokens, cs] contiguous -> rows of width ld at channel offset off
__global__ void k_copy_block(const float* __restrict__ s, int cs, float* __restrict__ d,
                             int ld, int off, int tokens) {
  int i = blockIdx.x * blockDim.x + threadIdx.x;
  if (i >= tokens * cs) return;
  int t = i / cs, c = i % cs;
  d[(size_t)t * ld + off + c] = s[i];
}

__global__ void k_glu(const float* __restrict__ in, float* __restrict__ out, int tokens) {
  int i = blockIdx.x * blockDim.x + threadIdx.x;
  if (i >= tokens * 128) return;
  int t = i / 128, c = i % 128;
  float a = in[(size_t)t * 256 + c];
  float g = in[(size_t)t * 256 + 128 + c];
  out[i] = a * sigmf(g);
}

// mode 0: alpha[channel];  mode 1: alpha[freq] (freq = (i/C)%Fdim)
__global__ void k_prelu(float* __restrict__ x, const float* __restrict__ al,
                        int mode, int C, int Fdim, int n) {
  int i = blockIdx.x * blockDim.x + threadIdx.x; if (i >= n) return;
  float v = x[i];
  float a = mode ? al[(i / C) % Fdim] : al[i % C];
  x[i] = (v >= 0.f) ? v : a * v;
}

__global__ void k_bnswish(float* __restrict__ x, const float* __restrict__ m,
                          const float* __restrict__ var, const float* __restrict__ g,
                          const float* __restrict__ b, int C, int n) {
  int i = blockIdx.x * blockDim.x + threadIdx.x; if (i >= n) return;
  int c = i % C;
  float v = (x[i] - m[c]) * rsqrtf(var[c] + 1e-5f) * g[c] + b[c];
  x[i] = v * sigmf(v);
}

// in [BT, F1, 128] -> out [BT, 2*F1, 64]; out[bt, 2*f1+j, cc] = in[bt, f1, j*64+cc]
__global__ void k_shuffle(const float* __restrict__ in, float* __restrict__ out,
                          int BT, int F1) {
  int i = blockIdx.x * blockDim.x + threadIdx.x;
  if (i >= BT * F1 * 128) return;
  int c = i % 128, f1 = (i / 128) % F1, bt = i / (128 * F1);
  int j = c / 64, cc = c % 64;
  out[((size_t)bt * (2 * F1) + 2 * f1 + j) * 64 + cc] = in[i];
}

__global__ void k_final(const float* __restrict__ xr, const float* __restrict__ xi,
                        const float* __restrict__ mag, const float* __restrict__ mask,
                        const float* __restrict__ cplx, float* __restrict__ out, int n) {
  int i = blockIdx.x * blockDim.x + threadIdx.x; if (i >= n) return;
  float m = mag[i];
  float rm = (m > 0.f) ? frcp(m) : 0.f;
  float cp = (m > 0.f) ? xr[i] * rm : 1.f;
  float sp = xi[i] * rm;
  float om = mask[i] * m;
  out[2 * i + 0] = om * cp + cplx[2 * i + 0];
  out[2 * i + 1] = om * sp + cplx[2 * i + 1];
}

__global__ void k_lnorm(const float* __restrict__ x, const float* __restrict__ g,
                        const float* __restrict__ b, float* __restrict__ out, int N, int C) {
  int t = blockIdx.x * blockDim.x + threadIdx.x; if (t >= N) return;
  const float* p = x + (size_t)t * C;
  float invC = frcp((float)C);
  float m = 0.f;
  for (int c = 0; c < C; ++c) m += p[c];
  m *= invC;
  float v = 0.f;
  for (int c = 0; c < C; ++c) { float d = p[c] - m; v += d * d; }
  v *= invC;
  float r = rsqrtf(v + 1e-5f);
  float* o = out + (size_t)t * C;
  for (int c = 0; c < C; ++c) o[c] = (p[c] - m) * r * g[c] + b[c];
}

// instance-norm stats: one block per (b,c); stats[bc] = {mean, rstd}
__global__ void k_inorm_reduce(const float* __restrict__ x, float* __restrict__ stats,
                               int Bn, int C, int SP) {
  int bc = blockIdx.x;
  int b = bc / C, c = bc % C;
  const float* p = x + ((size_t)b * SP) * C + c;
  float s = 0.f, ss = 0.f;
  for (int i = threadIdx.x; i < SP; i += blockDim.x) {
    float v = p[(size_t)i * C]; s += v; ss += v * v;
  }
  __shared__ float sh0[256], sh1[256];
  int tid = threadIdx.x;
  sh0[tid] = s; sh1[tid] = ss; __syncthreads();
  for (int off = 128; off > 0; off >>= 1) {
    if (tid < off) { sh0[tid] += sh0[tid + off]; sh1[tid] += sh1[tid + off]; }
    __syncthreads();
  }
  if (tid == 0) {
    float inv = frcp((float)SP);
    float m = sh0[0] * inv;
    float var = sh1[0] * inv - m * m;
    stats[2 * bc + 0] = m;
    stats[2 * bc + 1] = rsqrtf(var + 1e-5f);
  }
}

__global__ void k_inorm_prelu(float* __restrict__ x, const float* __restrict__ stats,
                              const float* __restrict__ g, const float* __restrict__ b,
                              const float* __restrict__ a, int C, int SP, int n) {
  int i = blockIdx.x * blockDim.x + threadIdx.x; if (i >= n) return;
  int c = i % C;
  int bidx = i / (SP * C);
  float m = stats[2 * (bidx * C + c) + 0];
  float r = stats[2 * (bidx * C + c) + 1];
  float v = (x[i] - m) * r * g[c] + b[c];
  x[i] = (v >= 0.f) ? v : a[c] * v;
}

// generic NHWC conv (VALID on implicitly zero-padded input) with dilated time,
// strided freq, row-stride ldin and optional reversed 64-channel block remap
// (revB = #blocks; ref concat order [newest..oldest] vs append storage).
__global__ void k_conv2d(const float* __restrict__ in, const float* __restrict__ w,
                         const float* __restrict__ bias, float* __restrict__ out,
                         int Bn, int Ti, int Fi, int Ci, int ldin, int revB,
                         int KT, int KF, int Co, int padT, int padF, int strF, int dilT,
                         int To, int Fo) {
  int gid = blockIdx.x * blockDim.x + threadIdx.x;
  int total = Bn * To * Fo * Co;
  if (gid >= total) return;
  int co = gid % Co;
  int f  = (gid / Co) % Fo;
  int t  = (gid / (Co * Fo)) % To;
  int b  = gid / (Co * Fo * To);
  float acc = bias ? bias[co] : 0.f;
  for (int kt = 0; kt < KT; ++kt) {
    int ti = t + kt * dilT - padT;
    if (ti < 0 || ti >= Ti) continue;
    for (int kf = 0; kf < KF; ++kf) {
      int fi = f * strF + kf - padF;
      if (fi < 0 || fi >= Fi) continue;
      const float* ip = in + ((size_t)(b * Ti + ti) * Fi + fi) * ldin;
      const float* wp = w + ((size_t)(kt * KF + kf) * Ci) * Co + co;
      if ((Ci & 63) == 0) {
        // storage-block order: contiguous input reads; remap weight block
        int nb = Ci >> 6;
        for (int blk = 0; blk < nb; ++blk) {
          int rb = revB ? (revB - 1 - blk) : blk;
          const float* ipb = ip + blk * 64;
          const float* wpb = wp + ((size_t)rb * 64) * Co;
          for (int cc = 0; cc < 64; ++cc)
            acc += ipb[cc] * wpb[(size_t)cc * Co];
        }
      } else {
        for (int ci = 0; ci < Ci; ++ci)
          acc += ip[ci] * wp[(size_t)ci * Co];
      }
    }
  }
  out[gid] = acc;
}

// depthwise 31-tap 'SAME' conv along the sequence axis of flat [tokens,C] data.
// token(s,p) = base(s) + p*strideTok,  base(s) = (s/modF)*mulDiv + (s%modF)*mulMod
__global__ void k_dwconv(const float* __restrict__ in, const float* __restrict__ w,
                         const float* __restrict__ db, float* __restrict__ out,
                         int nseq, int L, int C,
                         int modF, int mulDiv, int mulMod, int strideTok) {
  int gid = blockIdx.x * blockDim.x + threadIdx.x;
  int total = nseq * L * C;
  if (gid >= total) return;
  int c = gid % C;
  int r = gid / C;
  int p = r % L;
  int s = r / L;
  int base = (s / modF) * mulDiv + (s % modF) * mulMod;
  float acc = db[c];
  for (int k = 0; k < 31; ++k) {
    int pp = p + k - 15;
    if (pp < 0 || pp >= L) continue;
    acc += in[(size_t)(base + pp * strideTok) * C + c] * w[(size_t)k * C + c];
  }
  out[(size_t)(base + p * strideTok) * C + c] = acc;
}

// attention with relative position embedding, online softmax. heads=4, hdim=16.
// logit(i,j) = scale * sum_d q_d * (k_jd + pemb[i-j+512, d])
__global__ void k_attn(const float* __restrict__ q, const float* __restrict__ kv,
                       const float* __restrict__ pemb, float* __restrict__ ctx,
                       int nseq, int L, int modF, int mulDiv, int mulMod, int strideTok) {
  int gid = blockIdx.x * blockDim.x + threadIdx.x;
  int total = nseq * 4 * L;
  if (gid >= total) return;
  int i = gid % L;
  int sh = gid / L;
  int h = sh % 4;
  int s = sh / 4;
  int base = (s / modF) * mulDiv + (s % modF) * mulMod;
  int toki = base + i * strideTok;
  float qv[16];
#pragma unroll
  for (int d = 0; d < 16; ++d) qv[d] = q[(size_t)toki * 64 + h * 16 + d];
  const float scale = 0.25f; // 16^-0.5
  float mx = -1e30f, den = 0.f;
  float acc[16];
#pragma unroll
  for (int d = 0; d < 16; ++d) acc[d] = 0.f;
  for (int j = 0; j < L; ++j) {
    int tokj = base + j * strideTok;
    const float* kp = kv + (size_t)tokj * 128 + h * 16;
    const float* vp = kv + (size_t)tokj * 128 + 64 + h * 16;
    if (j + 1 < L)
      __builtin_prefetch(kv + (size_t)(base + (j + 1) * strideTok) * 128 + h * 16, 0, 1);
    int dd = i - j; dd = dd > 512 ? 512 : (dd < -512 ? -512 : dd);
    const float* rp = pemb + (size_t)(dd + 512) * 16;
    float sc = 0.f;
#pragma unroll
    for (int d = 0; d < 16; ++d) sc += qv[d] * (kp[d] + rp[d]);
    sc *= scale;
    float nm = fmaxf(mx, sc);
    float corr = __expf(mx - nm);
    float wv = __expf(sc - nm);
    den = den * corr + wv;
#pragma unroll
    for (int d = 0; d < 16; ++d) acc[d] = acc[d] * corr + wv * vp[d];
    mx = nm;
  }
  float inv = frcp(den);
#pragma unroll
  for (int d = 0; d < 16; ++d) ctx[(size_t)toki * 64 + h * 16 + d] = acc[d] * inv;
}

// ------------------- WMMA GEMM -------------------
// C[M,N] = act(A[M,K] * B[K,N] + bias). f32 storage, f16 multiply, f32
// accumulate. Requirements: K % 32 == 0, N % 64 == 0, A 16B-aligned.
// One wave computes a 16x64 tile: one A fragment vs 4 B fragments, 4 WMMAs
// per K-step. No load guards: out-of-range rows are clamped (their results
// are never stored), columns are always in range since N % 64 == 0.
__global__ void k_gemm_wmma(const float* __restrict__ A, const float* __restrict__ Bw,
                            const float* __restrict__ bias, float* __restrict__ C,
                            int M, int N, int K, int act) {
  int wid  = blockIdx.x * (blockDim.x >> 5) + (threadIdx.x >> 5);
  int lane = threadIdx.x & 31;
  int ngrp   = N >> 6;            // 64-col groups
  int mtiles = (M + 15) >> 4;
  if (wid >= ngrp * mtiles) return;
  int mt = wid / ngrp, ng = wid % ngrp;
  int m0 = mt << 4, n0 = ng << 6;
  int ml = lane & 15;   // A: local row; B/C: local col
  int h  = lane >> 4;   // half-wave select
  int mrow = m0 + ml; if (mrow > M - 1) mrow = M - 1;   // clamp (store-guarded)
  const float* Arow = A + (size_t)mrow * K;
  v8f acc0 = {}, acc1 = {}, acc2 = {}, acc3 = {};
  for (int kk = 0; kk < K; kk += 32) {
    // A fragment (16x32 f16): per lane two contiguous 8-float chunks.
    const float4* pa0 = (const float4*)(Arow + kk + h * 16);
    const float4* pa1 = (const float4*)(Arow + kk + 8 + h * 16);
    float4 x0 = pa0[0], x1 = pa0[1];
    float4 y0 = pa1[0], y1 = pa1[1];
    v16h a;
    a[0]  = (_Float16)x0.x; a[1]  = (_Float16)x0.y;
    a[2]  = (_Float16)x0.z; a[3]  = (_Float16)x0.w;
    a[4]  = (_Float16)x1.x; a[5]  = (_Float16)x1.y;
    a[6]  = (_Float16)x1.z; a[7]  = (_Float16)x1.w;
    a[8]  = (_Float16)y0.x; a[9]  = (_Float16)y0.y;
    a[10] = (_Float16)y0.z; a[11] = (_Float16)y0.w;
    a[12] = (_Float16)y1.x; a[13] = (_Float16)y1.y;
    a[14] = (_Float16)y1.z; a[15] = (_Float16)y1.w;
    // B fragments: lane holds column; element r = B[kk + 16h + r][col].
    const float* Bp = Bw + (size_t)(kk + 16 * h) * N + n0 + ml;
    v16h b0, b1, b2, b3;
#pragma unroll
    for (int r = 0; r < 16; ++r) {
      const float* row = Bp + (size_t)r * N;
      b0[r] = (_Float16)row[0];
      b1[r] = (_Float16)row[16];
      b2[r] = (_Float16)row[32];
      b3[r] = (_Float16)row[48];
    }
    acc0 = __builtin_amdgcn_wmma_f32_16x16x32_f16(false, a, false, b0, (short)0, acc0, false, false);
    acc1 = __builtin_amdgcn_wmma_f32_16x16x32_f16(false, a, false, b1, (short)0, acc1, false, false);
    acc2 = __builtin_amdgcn_wmma_f32_16x16x32_f16(false, a, false, b2, (short)0, acc2, false, false);
    acc3 = __builtin_amdgcn_wmma_f32_16x16x32_f16(false, a, false, b3, (short)0, acc3, false, false);
  }
  // epilogue: 4 col-tiles. Fast path (wave-uniform): full 16-row tile.
  if (m0 + 16 <= M) {
#pragma unroll
    for (int tile = 0; tile < 4; ++tile) {
      v8f acc = (tile == 0) ? acc0 : (tile == 1) ? acc1 : (tile == 2) ? acc2 : acc3;
      int ncol = n0 + tile * 16 + ml;
      float bv = bias ? bias[ncol] : 0.f;
#pragma unroll
      for (int r = 0; r < 8; ++r) {
        int mr = m0 + r + 8 * h;
        float val = acc[r] + bv;
        if (act == 1) val = val * sigmf(val); // swish
        C[(size_t)mr * N + ncol] = val;
      }
    }
  } else {
#pragma unroll
    for (int tile = 0; tile < 4; ++tile) {
      v8f acc = (tile == 0) ? acc0 : (tile == 1) ? acc1 : (tile == 2) ? acc2 : acc3;
      int ncol = n0 + tile * 16 + ml;
      float bv = bias ? bias[ncol] : 0.f;
#pragma unroll
      for (int r = 0; r < 8; ++r) {
        int mr = m0 + r + 8 * h;
        if (mr < M) {
          float val = acc[r] + bv;
          if (act == 1) val = val * sigmf(val);
          C[(size_t)mr * N + ncol] = val;
        }
      }
    }
  }
}

// =====================================================================
//                        HOST-SIDE STRUCTURE
// =====================================================================

struct DD  { const float *a[4], *b[4], *nb[4], *ng[4], *w[4]; };
struct FFp { const float *b1, *b2, *w1, *w2; };
struct Conf {
  const float *at_b, *at_g;
  const float *bo, *pemb, *wkv, *wo, *wq;                       // attn (sorted)
  const float *cv_b1, *cv_b2, *bn_b, *bn_g, *bn_m, *bn_v,
              *db, *dw, *lnb, *lng, *cv_w1, *cv_w2;             // conv (sorted)
  FFp ff1; const float *ff1_b, *ff1_g;
  FFp ff2; const float *ff2_b, *ff2_g;
  const float *n_b, *n_g;
};

static void gemm(hipStream_t st, const float* A, const float* B, const float* bias,
                 float* C, int M, int N, int K, int act) {
  // N % 64 == 0 and K % 32 == 0 for all projections in this model
  int tiles = cdiv(M, 16) * (N >> 6);
  k_gemm_wmma<<<cdiv(tiles, 8), 256, 0, st>>>(A, B, bias, C, M, N, K, act);
}

static void conv2d(hipStream_t st, const float* in, const float* w, const float* bias,
                   float* out, int Bn, int Ti, int Fi, int Ci, int ldin, int revB,
                   int KT, int KF, int Co, int padT, int padF, int strF, int dilT,
                   int To, int Fo) {
  int total = Bn * To * Fo * Co;
  k_conv2d<<<cdiv(total, 256), 256, 0, st>>>(in, w, bias, out, Bn, Ti, Fi, Ci, ldin,
                                             revB, KT, KF, Co, padT, padF, strF, dilT,
                                             To, Fo);
}

static void run_dildense(hipStream_t st, const float* x0, const DD& dd,
                         int Bn, int Ti, int Fi, float* cat, float* xi, float* stats) {
  int tokens = Bn * Ti * Fi;
  int n64 = tokens * 64;
  k_copy_block<<<cdiv(n64, 256), 256, 0, st>>>(x0, 64, cat, 256, 0, tokens);
  for (int i = 0; i < 4; ++i) {
    int d = 1 << i;
    int Ci = 64 * (i + 1);
    conv2d(st, cat, dd.w[i], dd.b[i], xi, Bn, Ti, Fi, Ci, 256, i + 1,
           2, 3, 64, /*padT=*/d, /*padF=*/1, /*strF=*/1, /*dilT=*/d, Ti, Fi);
    k_inorm_reduce<<<Bn * 64, 256, 0, st>>>(xi, stats, Bn, 64, Ti * Fi);
    k_inorm_prelu<<<cdiv(n64, 256), 256, 0, st>>>(xi, stats, dd.ng[i], dd.nb[i],
                                                  dd.a[i], 64, Ti * Fi, n64);
    if (i < 3)
      k_copy_block<<<cdiv(n64, 256), 256, 0, st>>>(xi, 64, cat, 256, (i + 1) * 64, tokens);
  }
  // result stays in xi
}

static void run_conformer(hipStream_t st, float* x, const Conf& c,
                          int nseq, int L, int modF, int mulDiv, int mulMod, int strideTok,
                          float* x0, float* bA, float* bB, float* bG, float* bD,
                          float* bQ, float* bKV, float* bC) {
  const int N = nseq * L;
  const int n64 = N * 64, n128 = N * 128;
  const int g64 = cdiv(n64, 256), g128 = cdiv(n128, 256), gN = cdiv(N, 256);
  k_copy<<<g64, 256, 0, st>>>(x0, x, n64);
  // ff1 (half-step)
  k_lnorm<<<gN, 256, 0, st>>>(x, c.ff1_g, c.ff1_b, bA, N, 64);
  gemm(st, bA, c.ff1.w1, c.ff1.b1, bB, N, 256, 64, 1);
  gemm(st, bB, c.ff1.w2, c.ff1.b2, bC, N, 64, 256, 0);
  k_axpy<<<g64, 256, 0, st>>>(x, bC, 0.5f, n64);
  // attention
  k_lnorm<<<gN, 256, 0, st>>>(x, c.at_g, c.at_b, bA, N, 64);
  gemm(st, bA, c.wq, nullptr, bQ, N, 64, 64, 0);
  gemm(st, bA, c.wkv, nullptr, bKV, N, 128, 64, 0);
  k_attn<<<cdiv(nseq * 4 * L, 256), 256, 0, st>>>(bQ, bKV, c.pemb, bC,
                                                  nseq, L, modF, mulDiv, mulMod, strideTok);
  gemm(st, bC, c.wo, c.bo, bA, N, 64, 64, 0);
  k_axpy<<<g64, 256, 0, st>>>(x, bA, 1.f, n64);
  // conv module
  k_lnorm<<<gN, 256, 0, st>>>(x, c.lng, c.lnb, bA, N, 64);
  gemm(st, bA, c.cv_w1, c.cv_b1, bB, N, 256, 64, 0);
  k_glu<<<g128, 256, 0, st>>>(bB, bG, N);
  k_dwconv<<<g128, 256, 0, st>>>(bG, c.dw, c.db, bD, nseq, L, 128,
                                 modF, mulDiv, mulMod, strideTok);
  k_bnswish<<<g128, 256, 0, st>>>(bD, c.bn_m, c.bn_v, c.bn_g, c.bn_b, 128, n128);
  gemm(st, bD, c.cv_w2, c.cv_b2, bA, N, 64, 128, 0);
  k_axpy<<<g64, 256, 0, st>>>(x, bA, 1.f, n64);
  // ff2 (half-step)
  k_lnorm<<<gN, 256, 0, st>>>(x, c.ff2_g, c.ff2_b, bA, N, 64);
  gemm(st, bA, c.ff2.w1, c.ff2.b1, bB, N, 256, 64, 1);
  gemm(st, bB, c.ff2.w2, c.ff2.b2, bC, N, 64, 256, 0);
  k_axpy<<<g64, 256, 0, st>>>(x, bC, 0.5f, n64);
  // final norm + outer residual (tscb's "+ xt")
  k_lnorm<<<gN, 256, 0, st>>>(x, c.n_g, c.n_b, bA, N, 64);
  k_add2<<<g64, 256, 0, st>>>(x, bA, x0, n64);
}

// =====================================================================
//                           kernel_launch
// =====================================================================

extern "C" void kernel_launch(void* const* d_in, const int* in_sizes, int n_in,
                              void* d_out, int out_size, void* d_ws, size_t ws_size,
                              hipStream_t stream) {
  (void)in_sizes; (void)n_in; (void)out_size; (void)ws_size;
  hipStream_t st = stream;
  const float* xr = (const float*)d_in[0];
  const float* xig = (const float*)d_in[1];
  int idx = 2;
  auto nxt = [&]() -> const float* { return (const float*)d_in[idx++]; };
  auto loadDD = [&](DD& d) {
    for (int i = 0; i < 4; ++i) d.a[i]  = nxt();
    for (int i = 0; i < 4; ++i) d.b[i]  = nxt();
    for (int i = 0; i < 4; ++i) d.nb[i] = nxt();
    for (int i = 0; i < 4; ++i) d.ng[i] = nxt();
    for (int i = 0; i < 4; ++i) d.w[i]  = nxt();
  };
  auto loadConf = [&](Conf& c) {
    c.at_b = nxt(); c.at_g = nxt();
    c.bo = nxt(); c.pemb = nxt(); c.wkv = nxt(); c.wo = nxt(); c.wq = nxt();
    c.cv_b1 = nxt(); c.cv_b2 = nxt(); c.bn_b = nxt(); c.bn_g = nxt();
    c.bn_m = nxt(); c.bn_v = nxt(); c.db = nxt(); c.dw = nxt();
    c.lnb = nxt(); c.lng = nxt(); c.cv_w1 = nxt(); c.cv_w2 = nxt();
    c.ff1.b1 = nxt(); c.ff1.b2 = nxt(); c.ff1.w1 = nxt(); c.ff1.w2 = nxt();
    c.ff1_b = nxt(); c.ff1_g = nxt();
    c.ff2.b1 = nxt(); c.ff2.b2 = nxt(); c.ff2.w1 = nxt(); c.ff2.w2 = nxt();
    c.ff2_b = nxt(); c.ff2_g = nxt();
    c.n_b = nxt(); c.n_g = nxt();
  };
  // top level sorted: cdec, denc, mdec, tscb
  struct { const float *a, *cb, *cw; DD dd; const float *nb, *ng, *sb, *sw; } cdec;
  cdec.a = nxt(); cdec.cb = nxt(); cdec.cw = nxt(); loadDD(cdec.dd);
  cdec.nb = nxt(); cdec.ng = nxt(); cdec.sb = nxt(); cdec.sw = nxt();

  struct { const float *a1, *a2, *b1, *b2; DD dd;
           const float *n1b, *n1g, *n2b, *n2g, *w1, *w2; } denc;
  denc.a1 = nxt(); denc.a2 = nxt(); denc.b1 = nxt(); denc.b2 = nxt(); loadDD(denc.dd);
  denc.n1b = nxt(); denc.n1g = nxt(); denc.n2b = nxt(); denc.n2g = nxt();
  denc.w1 = nxt(); denc.w2 = nxt();

  struct { const float *a, *a2, *c2b, *c2w, *cb, *cw; DD dd;
           const float *nb, *ng, *sb, *sw; } mdec;
  mdec.a = nxt(); mdec.a2 = nxt(); mdec.c2b = nxt(); mdec.c2w = nxt();
  mdec.cb = nxt(); mdec.cw = nxt(); loadDD(mdec.dd);
  mdec.nb = nxt(); mdec.ng = nxt(); mdec.sb = nxt(); mdec.sw = nxt();

  Conf tsf[4], tst[4];
  for (int i = 0; i < 4; ++i) { loadConf(tsf[i]); loadConf(tst[i]); } // keys: 'f' < 't'

  // ---------------- workspace carve (64-float = 256B aligned blocks) -----
  float* wsp = (float*)d_ws;
  size_t off = 0;
  auto alloc = [&](size_t n) {
    n = (n + 63) & ~(size_t)63;
    float* p = wsp + off; off += n; return p;
  };
  float* magbuf = alloc((size_t)N1);
  float* xin3   = alloc((size_t)N1 * 3);
  float* encA   = alloc((size_t)N1 * 64);
  float* cat    = alloc((size_t)N1 * 256);
  float* xi     = alloc((size_t)N1 * 64);
  float* stats  = alloc(512);
  float* xenc   = alloc((size_t)N2 * 64);
  float* x0b    = alloc((size_t)N2 * 64);
  float* bA     = alloc((size_t)N2 * 64);
  float* bB     = alloc((size_t)N2 * 256);
  float* bG     = alloc((size_t)N2 * 128);
  float* bD     = alloc((size_t)N2 * 128);
  float* bQ     = alloc((size_t)N2 * 64);
  float* bKV    = alloc((size_t)N2 * 128);
  float* bC     = alloc((size_t)N2 * 64);
  float* sub128 = alloc((size_t)N2 * 128);
  float* decA   = alloc((size_t)N3 * 64);
  float* m1     = alloc((size_t)N1);
  float* maskb  = alloc((size_t)N1);
  float* cplx   = alloc((size_t)N1 * 2);

  // ================= dense encoder =================
  k_prep<<<cdiv(N1, 256), 256, 0, st>>>(xr, xig, xin3, magbuf, N1);
  conv2d(st, xin3, denc.w1, denc.b1, encA, BN, TT, FQ, 3, 3, 0,
         1, 1, 64, 0, 0, 1, 1, TT, FQ);
  k_inorm_reduce<<<BN * 64, 256, 0, st>>>(encA, stats, BN, 64, TT * FQ);
  k_inorm_prelu<<<cdiv(N1 * 64, 256), 256, 0, st>>>(encA, stats, denc.n1g, denc.n1b,
                                                    denc.a1, 64, TT * FQ, N1 * 64);
  run_dildense(st, encA, denc.dd, BN, TT, FQ, cat, xi, stats);
  conv2d(st, xi, denc.w2, denc.b2, xenc, BN, TT, FQ, 64, 64, 0,
         1, 3, 64, 0, /*padF=*/1, /*strF=*/2, 1, TT, FE);
  k_inorm_reduce<<<BN * 64, 256, 0, st>>>(xenc, stats, BN, 64, TT * FE);
  k_inorm_prelu<<<cdiv(N2 * 64, 256), 256, 0, st>>>(xenc, stats, denc.n2g, denc.n2b,
                                                    denc.a2, 64, TT * FE, N2 * 64);

  // ================= 4x TSCB =================
  for (int i = 0; i < 4; ++i) {
    // time branch: seq s=(b,f): base = (s/FE)*(TT*FE) + (s%FE), stride = FE
    run_conformer(st, xenc, tst[i], BN * FE, TT, FE, TT * FE, 1, FE,
                  x0b, bA, bB, bG, bD, bQ, bKV, bC);
    // freq branch: seq s=(b,t): base = s*FE, stride = 1
    run_conformer(st, xenc, tsf[i], BN * TT, FE, BN * TT, 0, FE, 1,
                  x0b, bA, bB, bG, bD, bQ, bKV, bC);
  }

  // ================= mask decoder =================
  run_dildense(st, xenc, mdec.dd, BN, TT, FE, cat, xi, stats);
  conv2d(st, xi, mdec.sw, mdec.sb, sub128, BN, TT, FE, 64, 64, 0,
         1, 3, 128, 0, 1, 1, 1, TT, FE);
  k_shuffle<<<cdiv(N2 * 128, 256), 256, 0, st>>>(sub128, decA, BN * TT, FE);
  conv2d(st, decA, mdec.cw, mdec.cb, m1, BN, TT, F2, 64, 64, 0,
         1, 2, 1, 0, 0, 1, 1, TT, FQ);
  k_inorm_reduce<<<BN * 1, 256, 0, st>>>(m1, stats, BN, 1, TT * FQ);
  k_inorm_prelu<<<cdiv(N1, 256), 256, 0, st>>>(m1, stats, mdec.ng, mdec.nb,
                                               mdec.a, 1, TT * FQ, N1);
  conv2d(st, m1, mdec.c2w, mdec.c2b, maskb, BN, TT, FQ, 1, 1, 0,
         1, 1, 1, 0, 0, 1, 1, TT, FQ);
  k_prelu<<<cdiv(N1, 256), 256, 0, st>>>(maskb, mdec.a2, 1, 1, FQ, N1);

  // ================= complex decoder =================
  run_dildense(st, xenc, cdec.dd, BN, TT, FE, cat, xi, stats);
  conv2d(st, xi, cdec.sw, cdec.sb, sub128, BN, TT, FE, 64, 64, 0,
         1, 3, 128, 0, 1, 1, 1, TT, FE);
  k_shuffle<<<cdiv(N2 * 128, 256), 256, 0, st>>>(sub128, decA, BN * TT, FE);
  k_inorm_reduce<<<BN * 64, 256, 0, st>>>(decA, stats, BN, 64, TT * F2);
  k_inorm_prelu<<<cdiv(N3 * 64, 256), 256, 0, st>>>(decA, stats, cdec.ng, cdec.nb,
                                                    cdec.a, 64, TT * F2, N3 * 64);
  conv2d(st, decA, cdec.cw, cdec.cb, cplx, BN, TT, F2, 64, 64, 0,
         1, 2, 2, 0, 0, 1, 1, TT, FQ);

  // ================= combine =================
  k_final<<<cdiv(N1, 256), 256, 0, st>>>(xr, xig, magbuf, maskb, cplx,
                                         (float*)d_out, N1);
}
